// ThornsAttention_57397942944080
// MI455X (gfx1250) — matrix operand order repaired and checked by reading, not compiled, not run on hardware
//
#include <hip/hip_runtime.h>
#include <hip/hip_bf16.h>
#include <stdint.h>

// ---------------------------------------------------------------------------
// Problem constants (reference: B=2, S=2048, H=16, D=64, D_MODEL=1024)
// ---------------------------------------------------------------------------
#define BATCH   2
#define SEQ     2048
#define NHEADS  16
#define HDIM    64
#define DMODEL  1024
#define NROWS   (BATCH * SEQ)          // 4096

typedef __attribute__((ext_vector_type(16))) __bf16    v16bf;
typedef __attribute__((ext_vector_type(8)))  float     v8f;
typedef __attribute__((ext_vector_type(4)))  uint32_t  u32x4;

union Frag16 { v16bf v; u32x4 q[2]; };

__device__ __forceinline__ uint16_t f2bf(float f) {
    union { float f; uint32_t u; } c; c.f = f;
    uint32_t u = c.u + 0x7FFFu + ((c.u >> 16) & 1u);   // round-to-nearest-even
    return (uint16_t)(u >> 16);
}

__device__ __forceinline__ v8f vzero8() {
    v8f z;
#pragma unroll
    for (int i = 0; i < 8; ++i) z[i] = 0.0f;
    return z;
}

// CDNA5 async copy: global -> LDS, 16 bytes per lane, tracked by ASYNCcnt.
// (low 32 bits of a generic pointer derived from LDS == wave-relative LDS addr)
__device__ __forceinline__ void async_g2l_b128(const void* g, void* lds) {
    asm volatile("global_load_async_to_lds_b128 %0, %1, off"
                 :: "v"((uint32_t)(uintptr_t)lds), "v"(g)
                 : "memory");
}
#define WAIT_ASYNCCNT(n) asm volatile("s_wait_asynccnt " #n ::: "memory")

// ---------------------------------------------------------------------------
// fp32 -> bf16 conversion
// ---------------------------------------------------------------------------
__global__ void cvt_f32_bf16(const float* __restrict__ src,
                             uint16_t* __restrict__ dst, int n) {
    int i = blockIdx.x * blockDim.x + threadIdx.x;
    int stride = gridDim.x * blockDim.x;
    for (; i < n; i += stride) dst[i] = f2bf(src[i]);
}

// ---------------------------------------------------------------------------
// Tiled GEMM: C[M,N] = A[M,K] * Bw[N,K]^T   (A,Bw bf16 row-major, f32 accum)
// mode 0: Cf fp32 row-major [M,N]      (final projection -> d_out)
// mode 1: Cb bf16 in [B,H,S,D] layout  (Q/K/V projections)
// Block: 256 threads = 8 waves, each wave owns a 32x64 sub-tile (2x4 WMMA).
// Double-buffered LDS staged with GLOBAL_LOAD_ASYNC_TO_LDS_B128; last tile
// peeled so the steady-state loop is branch-free.
// ---------------------------------------------------------------------------
#define BM 128
#define BN 128
#define BK 32
#define LPAD 40  // ushort stride (80B rows, 16B aligned)

__global__ __launch_bounds__(256)
void gemm_bf16_wmma(const uint16_t* __restrict__ A,
                    const uint16_t* __restrict__ Bw,
                    float* __restrict__ Cf, uint16_t* __restrict__ Cb,
                    int M, int N, int K, int mode) {
    __shared__ __align__(16) uint16_t As[2][BM][LPAD];
    __shared__ __align__(16) uint16_t Bs[2][BM][LPAD];

    const int tid  = threadIdx.x;
    const int w    = tid >> 5;
    const int lane = tid & 31;
    const int half = lane >> 4;
    const int n16  = lane & 15;
    const int wm   = w & 3;     // 4 wave-rows of 32
    const int wn   = w >> 2;    // 2 wave-cols of 64
    const int rowA0 = blockIdx.y * BM;
    const int rowB0 = blockIdx.x * BN;

    // each thread stages 2x16B of A and 2x16B of B per tile (4 async ops)
    const int sr0 = tid >> 2;              // rows 0..63
    const int sr1 = sr0 + 64;              // rows 64..127
    const int sc  = (tid & 3) << 3;        // bf16 col 0/8/16/24

    auto issue_tile = [&](int buf, int kb) {
        async_g2l_b128(A  + (size_t)(rowA0 + sr0) * K + kb + sc, &As[buf][sr0][sc]);
        async_g2l_b128(A  + (size_t)(rowA0 + sr1) * K + kb + sc, &As[buf][sr1][sc]);
        async_g2l_b128(Bw + (size_t)(rowB0 + sr0) * K + kb + sc, &Bs[buf][sr0][sc]);
        async_g2l_b128(Bw + (size_t)(rowB0 + sr1) * K + kb + sc, &Bs[buf][sr1][sc]);
    };

    v8f acc[2][4];
#pragma unroll
    for (int mt = 0; mt < 2; ++mt)
#pragma unroll
        for (int nt = 0; nt < 4; ++nt) acc[mt][nt] = vzero8();

    auto do_tile = [&](int buf) {
        Frag16 af[2], bfr[4];
#pragma unroll
        for (int mt = 0; mt < 2; ++mt) {
            int r = wm * 32 + mt * 16 + n16;
            // A layout: k = e + 8*half (e<8) ; k = 16 + (e-8) + 8*half (e>=8)
            af[mt].q[0] = *(const u32x4*)&As[buf][r][8 * half];
            af[mt].q[1] = *(const u32x4*)&As[buf][r][16 + 8 * half];
        }
#pragma unroll
        for (int nt = 0; nt < 4; ++nt) {
            int r = wn * 64 + nt * 16 + n16;
            // B layout: k = e + 8*half (e<8) ; k = 16 + (e-8) + 8*half (e>=8)
            bfr[nt].q[0] = *(const u32x4*)&Bs[buf][r][8 * half];
            bfr[nt].q[1] = *(const u32x4*)&Bs[buf][r][16 + 8 * half];
        }
#pragma unroll
        for (int mt = 0; mt < 2; ++mt)
#pragma unroll
            for (int nt = 0; nt < 4; ++nt)
                acc[mt][nt] = __builtin_amdgcn_wmma_f32_16x16x32_bf16(
                    false, af[mt].v, false, bfr[nt].v,
                    (short)0, acc[mt][nt], false, false);
    };

    const int ntiles = K / BK;
    issue_tile(0, 0);
    int cur = 0;
    for (int t = 0; t < ntiles - 1; ++t) {       // branch-free steady state
        issue_tile(cur ^ 1, (t + 1) * BK);       // prefetch next tile
        WAIT_ASYNCCNT(0x4);                      // tile t landed (in-order)
        __syncthreads();                         // visible to all waves
        do_tile(cur);
        __syncthreads();                         // reads done before overwrite
        cur ^= 1;
    }
    WAIT_ASYNCCNT(0x0);                          // peeled final tile
    __syncthreads();
    do_tile(cur);

    // Epilogue: C layout m = e + 8*half, n = lane&15
#pragma unroll
    for (int mt = 0; mt < 2; ++mt)
#pragma unroll
        for (int nt = 0; nt < 4; ++nt)
#pragma unroll
            for (int e = 0; e < 8; ++e) {
                int row = rowA0 + wm * 32 + mt * 16 + e + 8 * half;
                int col = rowB0 + wn * 64 + nt * 16 + n16;
                float v = acc[mt][nt][e];
                if (mode == 0) {
                    Cf[(size_t)row * N + col] = v;
                } else {
                    int b = row >> 11, s = row & (SEQ - 1);
                    int h = col >> 6,  d = col & (HDIM - 1);
                    Cb[((((size_t)(b * NHEADS + h) << 11) | s) << 6) | d] = f2bf(v);
                }
            }
}

// ---------------------------------------------------------------------------
// Flash attention with ALiBi + causal mask.
// Grid: (S/64, H, B). Block: 128 threads = 4 waves; wave w owns q rows
// [qb*64 + w*16, +16). Online softmax in fp32; QK^T and PV on WMMA bf16.
// K staged via async-to-LDS, V staged transposed; both double-buffered so
// the next block's staging overlaps this block's WMMA + softmax work.
// ---------------------------------------------------------------------------
__global__ __launch_bounds__(128)
void attn_alibi_wmma(const uint16_t* __restrict__ Q,
                     const uint16_t* __restrict__ Km,
                     const uint16_t* __restrict__ Vm,
                     uint16_t* __restrict__ O) {
    __shared__ __align__(16) uint16_t Ks[2][32][72];   // K tile  [kpos][d]
    __shared__ __align__(16) uint16_t Vt[2][64][40];   // V tile transposed [d][kpos]
    __shared__ __align__(16) uint16_t Pl[4][16][32];   // per-wave P staging

    const int tid  = threadIdx.x;
    const int w    = tid >> 5;
    const int lane = tid & 31;
    const int half = lane >> 4;
    const int n16  = lane & 15;
    const int b  = blockIdx.z;
    const int h  = blockIdx.y;
    const int qb = blockIdx.x;
    const int qbase = qb * 64 + w * 16;
    const size_t headOff = (size_t)(b * NHEADS + h) * SEQ * HDIM;
    const float slope = exp2f(-0.5f * (float)(h + 1));  // (2^8)^(1/16) slopes
    const float scale = 0.125f;                          // 1/sqrt(64)

    // Q fragments (A operand), kept in registers for the whole loop
    Frag16 qf[2];
    {
        const uint16_t* qrow = Q + headOff + (size_t)(qbase + n16) * HDIM;
#pragma unroll
        for (int c = 0; c < 2; ++c) {
            qf[c].q[0] = *(const u32x4*)(qrow + 32 * c + 8 * half);
            qf[c].q[1] = *(const u32x4*)(qrow + 32 * c + 16 + 8 * half);
        }
    }

    float mi[8], li[8];
    v8f accO[4];
#pragma unroll
    for (int e = 0; e < 8; ++e) { mi[e] = -1e30f; li[e] = 0.0f; }
#pragma unroll
    for (int nt = 0; nt < 4; ++nt) accO[nt] = vzero8();

    // Stage one 32x64 K/V block: K via async copy, V transposed manually
    auto stage = [&](int buf, int kb) {
#pragma unroll
        for (int p = 0; p < 2; ++p) {
            int idx = tid + p * 128;
            int r   = idx >> 3;
            int c8  = (idx & 7) << 3;
            async_g2l_b128(Km + headOff + (size_t)(kb + r) * HDIM + c8,
                           &Ks[buf][r][c8]);
            u32x4 vv = *(const u32x4*)(Vm + headOff + (size_t)(kb + r) * HDIM + c8);
            const uint16_t* vs = (const uint16_t*)&vv;
#pragma unroll
            for (int j = 0; j < 8; ++j) Vt[buf][c8 + j][r] = vs[j];
        }
    };

    // One 32-wide k-block: QK^T WMMAs, online softmax, PV WMMAs
    auto process = [&](int buf, int kb) {
        v8f sc[2];
#pragma unroll
        for (int nt = 0; nt < 2; ++nt) {
            sc[nt] = vzero8();
#pragma unroll
            for (int c = 0; c < 2; ++c) {
                Frag16 kf;
                kf.q[0] = *(const u32x4*)&Ks[buf][nt * 16 + n16][32 * c + 8 * half];
                kf.q[1] = *(const u32x4*)&Ks[buf][nt * 16 + n16][32 * c + 16 + 8 * half];
                sc[nt] = __builtin_amdgcn_wmma_f32_16x16x32_bf16(
                    false, qf[c].v, false, kf.v, (short)0, sc[nt], false, false);
            }
        }
        // Online softmax (rows distributed as m = e + 8*half; n over 16 lanes)
#pragma unroll
        for (int e = 0; e < 8; ++e) {
            int qr = qbase + e + 8 * half;
            int kL = kb + n16;
            int kH = kb + 16 + n16;
            float a0 = sc[0][e] * scale - slope * (float)(kL - qr);
            float a1 = sc[1][e] * scale - slope * (float)(kH - qr);
            if (kL > qr) a0 = -1e30f;
            if (kH > qr) a1 = -1e30f;
            float mx = fmaxf(a0, a1);
#pragma unroll
            for (int off = 1; off < 16; off <<= 1)
                mx = fmaxf(mx, __shfl_xor(mx, off, 32));
            float mnew  = fmaxf(mi[e], mx);
            float alpha = __expf(mi[e] - mnew);
            float p0 = __expf(a0 - mnew);
            float p1 = __expf(a1 - mnew);
            float ps = p0 + p1;
#pragma unroll
            for (int off = 1; off < 16; off <<= 1)
                ps += __shfl_xor(ps, off, 32);
            li[e] = li[e] * alpha + ps;
            mi[e] = mnew;
#pragma unroll
            for (int nt = 0; nt < 4; ++nt) accO[nt][e] *= alpha;
            // stage P (C layout) into LDS for A-layout reload
            Pl[w][e + 8 * half][n16]      = f2bf(p0);
            Pl[w][e + 8 * half][16 + n16] = f2bf(p1);
        }
        // Reload P as A operand (row = lane&15, k = A-layout map)
        Frag16 pf;
        pf.q[0] = *(const u32x4*)&Pl[w][n16][8 * half];
        pf.q[1] = *(const u32x4*)&Pl[w][n16][16 + 8 * half];
        // PV: B operand from transposed V (lane = d col, k = A-layout map)
#pragma unroll
        for (int nt = 0; nt < 4; ++nt) {
            Frag16 vf;
            vf.q[0] = *(const u32x4*)&Vt[buf][nt * 16 + n16][8 * half];
            vf.q[1] = *(const u32x4*)&Vt[buf][nt * 16 + n16][16 + 8 * half];
            accO[nt] = __builtin_amdgcn_wmma_f32_16x16x32_bf16(
                false, pf.v, false, vf.v, (short)0, accO[nt], false, false);
        }
    };

    const int nblocks = 2 * qb + 2;    // kEnd/32, >= 2
    stage(0, 0);
    int cur = 0;
    for (int t = 0; t < nblocks - 1; ++t) {
        stage(cur ^ 1, (t + 1) * 32);  // prefetch next k-block
        WAIT_ASYNCCNT(0x2);            // block t's K landed (in-order)
        __syncthreads();
        process(cur, t * 32);
        __syncthreads();               // buf reads done before overwrite
        cur ^= 1;
    }
    WAIT_ASYNCCNT(0x0);                // peeled final block
    __syncthreads();
    process(cur, (nblocks - 1) * 32);

    // Write O in [B, S, H*D] bf16 layout (row-major input to final GEMM)
#pragma unroll
    for (int nt = 0; nt < 4; ++nt)
#pragma unroll
        for (int e = 0; e < 8; ++e) {
            int qr = qbase + e + 8 * half;
            float v = accO[nt][e] * (1.0f / li[e]);
            O[(size_t)(b * SEQ + qr) * DMODEL + h * HDIM + nt * 16 + n16] = f2bf(v);
        }
}

// ---------------------------------------------------------------------------
// Launch
// ---------------------------------------------------------------------------
extern "C" void kernel_launch(void* const* d_in, const int* in_sizes, int n_in,
                              void* d_out, int out_size, void* d_ws, size_t ws_size,
                              hipStream_t stream) {
    (void)in_sizes; (void)n_in; (void)out_size; (void)ws_size;
    const float* x  = (const float*)d_in[0];
    const float* Wq = (const float*)d_in[1];
    const float* Wk = (const float*)d_in[2];
    const float* Wv = (const float*)d_in[3];
    const float* Wd = (const float*)d_in[4];

    uint8_t* ws = (uint8_t*)d_ws;
    const size_t MB = 1024 * 1024;
    uint16_t* xb  = (uint16_t*)(ws + 0 * MB);   // 8 MB  [4096,1024]
    uint16_t* wqb = (uint16_t*)(ws + 8 * MB);   // 2 MB
    uint16_t* wkb = (uint16_t*)(ws + 10 * MB);
    uint16_t* wvb = (uint16_t*)(ws + 12 * MB);
    uint16_t* wdb = (uint16_t*)(ws + 14 * MB);
    uint16_t* Qb  = (uint16_t*)(ws + 16 * MB);  // 8 MB  [B,H,S,D]
    uint16_t* Kb  = (uint16_t*)(ws + 24 * MB);
    uint16_t* Vb  = (uint16_t*)(ws + 32 * MB);
    uint16_t* Ob  = (uint16_t*)(ws + 40 * MB);  // 8 MB  [4096,1024]

    const int nX = NROWS * DMODEL;     // 4M
    const int nW = DMODEL * DMODEL;    // 1M
    cvt_f32_bf16<<<1024, 256, 0, stream>>>(x,  xb,  nX);
    cvt_f32_bf16<<<512,  256, 0, stream>>>(Wq, wqb, nW);
    cvt_f32_bf16<<<512,  256, 0, stream>>>(Wk, wkb, nW);
    cvt_f32_bf16<<<512,  256, 0, stream>>>(Wv, wvb, nW);
    cvt_f32_bf16<<<512,  256, 0, stream>>>(Wd, wdb, nW);

    dim3 gg(DMODEL / BN, NROWS / BM);  // (8, 32)
    gemm_bf16_wmma<<<gg, 256, 0, stream>>>(xb, wqb, nullptr, Qb,
                                           NROWS, DMODEL, DMODEL, 1);
    gemm_bf16_wmma<<<gg, 256, 0, stream>>>(xb, wkb, nullptr, Kb,
                                           NROWS, DMODEL, DMODEL, 1);
    gemm_bf16_wmma<<<gg, 256, 0, stream>>>(xb, wvb, nullptr, Vb,
                                           NROWS, DMODEL, DMODEL, 1);

    attn_alibi_wmma<<<dim3(SEQ / 64, NHEADS, BATCH), 128, 0, stream>>>(Qb, Kb, Vb, Ob);

    gemm_bf16_wmma<<<gg, 256, 0, stream>>>(Ob, wdb, (float*)d_out, nullptr,
                                           NROWS, DMODEL, DMODEL, 0);
}
